// CustomBertSelfAttention_37538014167339
// MI455X (gfx1250) — matrix operand order repaired and checked by reading, not compiled, not run on hardware
//
#include <hip/hip_runtime.h>
#include <hip/hip_bf16.h>
#include <stdint.h>

// Problem constants (from the reference)
#define BB 2
#define SS 2048
#define DD 1024
#define HH 16
#define HD 64
#define NN (BB * SS)   // 4096 total rows of hidden_states

typedef __attribute__((ext_vector_type(16))) _Float16 v16h;
typedef __attribute__((ext_vector_type(8)))  _Float16 v8h;
typedef __attribute__((ext_vector_type(4)))  _Float16 v4h;
typedef __attribute__((ext_vector_type(8)))  float    v8f;

// ---------------------------------------------------------------------------
// WMMA helper: D = A(16x32 f16) * B(32x16 f16) + C(16x16 f32)
// ---------------------------------------------------------------------------
static __device__ __forceinline__ v8f wmma_f16(v16h a, v16h b, v8f c) {
  return __builtin_amdgcn_wmma_f32_16x16x32_f16(false, a, false, b, (short)0, c,
                                                false, false);
}

static __device__ __forceinline__ v16h load16(const _Float16* p) {
  return *reinterpret_cast<const v16h*>(p);
}

// A-fragment (16x32, f16), contiguous k order (same as the B loads):
//   lanes 0-15 : M = lane,    halves[0..15] = K k0+0..15
//   lanes 16-31: M = lane-16, halves[0..15] = K k0+16..31
static __device__ __forceinline__ v16h load_a_frag(const _Float16* base, int row0,
                                                   int lda, int k0) {
  const int lane = threadIdx.x & 31;
  const int m    = lane & 15;
  const int kg   = (lane >> 4) << 4;  // 0 or 16
  const _Float16* p = base + (size_t)(row0 + m) * lda + k0 + kg;
  v8h lo = *reinterpret_cast<const v8h*>(p);
  v8h hi = *reinterpret_cast<const v8h*>(p + 8);
  v16h r;
#pragma unroll
  for (int i = 0; i < 8; ++i) { r[i] = lo[i]; r[i + 8] = hi[i]; }
  return r;
}

// Row reductions across the 16 lanes that hold one C-fragment row
static __device__ __forceinline__ float redmax16(float v) {
#pragma unroll
  for (int m = 8; m >= 1; m >>= 1) v = fmaxf(v, __shfl_xor(v, m, 32));
  return v;
}
static __device__ __forceinline__ float redsum16(float v) {
#pragma unroll
  for (int m = 8; m >= 1; m >>= 1) v += __shfl_xor(v, m, 32);
  return v;
}

// ---------------------------------------------------------------------------
// CDNA5 async copy: 16B per lane, global -> LDS, tracked by ASYNCcnt.
// Inline asm for toolchain portability (bridge doc: builtin arity differs).
// ---------------------------------------------------------------------------
static __device__ __forceinline__ void async_copy_b128(void* lds_dst,
                                                       const void* gsrc) {
  const uint32_t loff = (uint32_t)(uintptr_t)lds_dst;       // LDS byte address
  const uint64_t ga   = (uint64_t)(uintptr_t)gsrc;          // global address
  asm volatile("global_load_async_to_lds_b128 %0, %1, off"
               :: "v"(loff), "v"(ga) : "memory");
}

// Wait until at most `n` async ops remain outstanding (immediate).
#if defined(__has_builtin) && __has_builtin(__builtin_amdgcn_s_wait_asynccnt)
#define WAIT_ASYNC(n) __builtin_amdgcn_s_wait_asynccnt((n))
#else
#define WAIT_ASYNC(n) asm volatile("s_wait_asynccnt %0" ::"i"(n) : "memory")
#endif

// ---------------------------------------------------------------------------
// Phase 0a: fp32 -> f16 convert of hidden_states
// ---------------------------------------------------------------------------
__global__ __launch_bounds__(256) void k_xcvt(const float* __restrict__ X,
                                              _Float16* __restrict__ Xh) {
  const size_t i = ((size_t)blockIdx.x * blockDim.x + threadIdx.x) * 4;
  float4 f = *reinterpret_cast<const float4*>(X + i);
  v4h h;
  h[0] = (_Float16)f.x; h[1] = (_Float16)f.y;
  h[2] = (_Float16)f.z; h[3] = (_Float16)f.w;
  *reinterpret_cast<v4h*>(Xh + i) = h;
}

// ---------------------------------------------------------------------------
// Phase 0b: W[k][n] fp32 -> WT[n][k] f16 (LDS-tiled transpose), z selects q/k/v
// ---------------------------------------------------------------------------
__global__ __launch_bounds__(256) void k_wtrans(
    const float* __restrict__ Wq, const float* __restrict__ Wk,
    const float* __restrict__ Wv, _Float16* __restrict__ WTq,
    _Float16* __restrict__ WTk, _Float16* __restrict__ WTv) {
  __shared__ float tile[32][33];  // +1 pad: avoid LDS bank conflicts
  const int z = blockIdx.z;
  const float* W  = (z == 0) ? Wq : (z == 1) ? Wk : Wv;
  _Float16*   WT  = (z == 0) ? WTq : (z == 1) ? WTk : WTv;
  const int k0 = blockIdx.x * 32;
  const int n0 = blockIdx.y * 32;
  const int tx = threadIdx.x & 31;
  const int ty = threadIdx.x >> 5;  // 0..7
#pragma unroll
  for (int i = 0; i < 32; i += 8)
    tile[ty + i][tx] = W[(size_t)(k0 + ty + i) * DD + n0 + tx];
  __syncthreads();
#pragma unroll
  for (int i = 0; i < 32; i += 8)
    WT[(size_t)(n0 + ty + i) * DD + k0 + tx] = (_Float16)tile[tx][ty + i];
}

// ---------------------------------------------------------------------------
// Phase 1: fused QKV projection GEMM, WMMA 16x16x32 f16 -> f32
//   grid = (NN/32, HH, 3), block = 32 (one wave = one 32x64 output tile:
//   2 A-fragments x 4 B-fragments -> 8 WMMAs per K-step, 2x B reuse)
//   z==0: Q (scaled by 1/sqrt(HD)) stored [bh][s][d]
//   z==1: K stored [bh][s][d]   (row-major == B-fragment layout for Q*K^T)
//   z==2: V stored [bh][d][s]   (transposed == B-fragment layout for P*V)
// ---------------------------------------------------------------------------
__global__ __launch_bounds__(32) void k_qkv_gemm(
    const _Float16* __restrict__ Xh, const _Float16* __restrict__ WTq,
    const _Float16* __restrict__ WTk, const _Float16* __restrict__ WTv,
    const float* __restrict__ bq, const float* __restrict__ bk,
    const float* __restrict__ bv, _Float16* __restrict__ Qh,
    _Float16* __restrict__ Kh, _Float16* __restrict__ Vt) {
  const int z = blockIdx.z;
  const _Float16* WT = (z == 0) ? WTq : (z == 1) ? WTk : WTv;
  const float* bias  = (z == 0) ? bq : (z == 1) ? bk : bv;

  const int lane = threadIdx.x & 31;
  const int nlo  = lane & 15;
  const int hi16 = (lane >> 4) << 4;
  const int hi8g = (lane >> 4) << 3;
  const int r0   = blockIdx.x * 32;   // 32-row tile in N
  const int h    = blockIdx.y;        // 64-column tile == head
  const int col0 = h * HD;

  v8f acc[2][4] = {};
#pragma unroll 1
  for (int k0 = 0; k0 < DD; k0 += 32) {
    v16h a0 = load_a_frag(Xh, r0,      DD, k0);
    v16h a1 = load_a_frag(Xh, r0 + 16, DD, k0);
#pragma unroll
    for (int t = 0; t < 4; ++t) {
      const _Float16* bp = WT + (size_t)(col0 + t * 16 + nlo) * DD + k0 + hi16;
      v16h bfr = load16(bp);
      acc[0][t] = wmma_f16(a0, bfr, acc[0][t]);
      acc[1][t] = wmma_f16(a1, bfr, acc[1][t]);
    }
  }
#pragma unroll
  for (int mtile = 0; mtile < 2; ++mtile) {
#pragma unroll
    for (int t = 0; t < 4; ++t) {
      const int d    = t * 16 + nlo;
      const float bb = bias[col0 + d];
#pragma unroll
      for (int r = 0; r < 8; ++r) {
        const int row = r0 + mtile * 16 + r + hi8g;  // global row in N
        const int b   = row >> 11;                   // / SS
        const int s   = row & (SS - 1);
        const size_t bh = (size_t)b * HH + h;
        float v = acc[mtile][t][r] + bb;
        if (z == 0)      Qh[(bh * SS + s) * HD + d] = (_Float16)(v * 0.125f);
        else if (z == 1) Kh[(bh * SS + s) * HD + d] = (_Float16)v;
        else             Vt[(bh * HD + d) * SS + s] = (_Float16)v;
      }
    }
  }
}

// ---------------------------------------------------------------------------
// Phase 2: flash attention.  grid = (SS/64, HH, BB), block = 128 (4 waves).
// Double-buffered CDNA5 async-to-LDS staging of 32-key K/V tiles: the fill
// for tile i+1 streams (ASYNCcnt) while tile i is consumed by WMMA.  Each
// thread issues exactly 4 async b128 ops per tile and async loads complete
// in order, so `s_wait_asynccnt 4` releases the older tile's fill.
// ---------------------------------------------------------------------------
__global__ __launch_bounds__(128) void k_attn(
    const _Float16* __restrict__ Qh, const _Float16* __restrict__ Kh,
    const _Float16* __restrict__ Vt, const float* __restrict__ bias,
    const float* __restrict__ coef, const float* __restrict__ mask,
    float* __restrict__ out) {
  __shared__ __align__(16) _Float16 kt[2][32 * 64];    // [buf][key][d]  8 KB
  __shared__ __align__(16) _Float16 vt[2][64 * 32];    // [buf][d][key]  8 KB
  __shared__ __align__(16) _Float16 ptile[4][16 * 32]; // per-wave P     4 KB

  const int tid  = threadIdx.x;       // 0..127
  const int lane = tid & 31;
  const int wid  = tid >> 5;          // wave id 0..3
  const int nlo  = lane & 15;
  const int hig  = lane >> 4;
  const int hi8g = hig << 3;
  const int hi16 = hig << 4;

  const int q0 = blockIdx.x * 64 + wid * 16;  // this wave's 16-query tile
  const int h  = blockIdx.y;
  const int b  = blockIdx.z;
  const size_t bh = (size_t)b * HH + h;

  const _Float16* Qb = Qh + bh * (size_t)SS * HD;
  const _Float16* Kb = Kh + bh * (size_t)SS * HD;
  const _Float16* Vb = Vt + bh * (size_t)HD * SS;
  const float c = coef[0];

  // Cooperative async fill of one 32-key K/V tile pair (4 b128 ops/thread).
  auto fill_tiles = [&](int buf, int kc) {
#pragma unroll
    for (int cc = 0; cc < 2; ++cc) {
      const int ch  = tid + cc * 128;          // 0..255
      const int key = ch >> 3, dp = (ch & 7) * 8;
      async_copy_b128(&kt[buf][key * 64 + dp],
                      Kb + (size_t)(kc + key) * HD + dp);
      const int dv = ch >> 2, kp = (ch & 3) * 8;
      async_copy_b128(&vt[buf][dv * 32 + kp], Vb + (size_t)dv * SS + kc + kp);
    }
  };

  // Q fragments (Q pre-scaled by 1/sqrt(HD)): A = 16 x 64 over two K=32 steps
  v16h qa0 = load_a_frag(Qb, q0, HD, 0);
  v16h qa1 = load_a_frag(Qb, q0, HD, 32);

  v8f acc[4] = {};
  float m_i[8], l_i[8];
#pragma unroll
  for (int r = 0; r < 8; ++r) { m_i[r] = -3.0e30f; l_i[r] = 0.0f; }

  fill_tiles(0, 0);  // prologue fill of buffer 0

#pragma unroll 1
  for (int kc = 0; kc < SS; kc += 32) {
    const int cur = (kc >> 5) & 1;
    if (kc + 32 < SS) {
      fill_tiles(cur ^ 1, kc + 32);  // stream next tile in the background
      WAIT_ASYNC(4);                 // release current tile (older 4 ops/thread)
    } else {
      WAIT_ASYNC(0);
    }
    __syncthreads();

    __builtin_prefetch(bias + (size_t)(q0 + hi8g) * SS + kc + 64, 0, 0);

    // ---- scores: Q(16x64) * K^T(64x32) from LDS ----
    v8f sc[2];
#pragma unroll
    for (int sub = 0; sub < 2; ++sub) {
      const int kk = sub * 16;  // key offset within tile
      const _Float16* kp = &kt[cur][(kk + nlo) * 64];
      v8f z = {};
      v8f s0 = wmma_f16(qa0, load16(kp + hi16), z);
      s0 = wmma_f16(qa1, load16(kp + 32 + hi16), s0);
      const float mk = mask[(size_t)b * SS + kc + kk + nlo];
#pragma unroll
      for (int r = 0; r < 8; ++r) {
        const int row = q0 + r + hi8g;
        s0[r] += bias[(size_t)row * SS + kc + kk + nlo] * c + mk;
      }
      sc[sub] = s0;
    }
    // ---- online softmax over the 32 new keys ----
#pragma unroll
    for (int r = 0; r < 8; ++r) {
      float mx   = redmax16(fmaxf(sc[0][r], sc[1][r]));
      float mnew = fmaxf(m_i[r], mx);
      float p0   = __expf(sc[0][r] - mnew);
      float p1   = __expf(sc[1][r] - mnew);
      float rs   = redsum16(p0 + p1);
      float corr = __expf(m_i[r] - mnew);
      l_i[r] = l_i[r] * corr + rs;
      m_i[r] = mnew;
      sc[0][r] = p0; sc[1][r] = p1;
#pragma unroll
      for (int t = 0; t < 4; ++t) acc[t][r] *= corr;
    }
    // ---- C-fragment -> A-fragment re-layout of P via per-wave LDS tile ----
    _Float16* pt = ptile[wid];
#pragma unroll
    for (int r = 0; r < 8; ++r) {
      const int row = r + hi8g;
      pt[row * 32 + nlo]      = (_Float16)sc[0][r];
      pt[row * 32 + 16 + nlo] = (_Float16)sc[1][r];
    }
    v16h pa;
    {
      const _Float16* pp = &pt[nlo * 32 + hi16];
      v8h lo = *reinterpret_cast<const v8h*>(pp);
      v8h hi = *reinterpret_cast<const v8h*>(pp + 8);
#pragma unroll
      for (int i = 0; i < 8; ++i) { pa[i] = lo[i]; pa[i + 8] = hi[i]; }
    }
    // ---- ctx += P(16x32) * V(32x64) from LDS ----
#pragma unroll
    for (int t = 0; t < 4; ++t) {
      const _Float16* vp = &vt[cur][(t * 16 + nlo) * 32 + hi16];
      acc[t] = wmma_f16(pa, load16(vp), acc[t]);
    }
    __syncthreads();  // all waves done with buf `cur` before it is refilled
  }

  // epilogue: normalize by row sums, write [B,S,D] fp32
#pragma unroll
  for (int t = 0; t < 4; ++t) {
#pragma unroll
    for (int r = 0; r < 8; ++r) {
      const int row = q0 + r + hi8g;
      out[((size_t)b * SS + row) * DD + h * HD + t * 16 + nlo] =
          acc[t][r] / l_i[r];
    }
  }
}

// ---------------------------------------------------------------------------
extern "C" void kernel_launch(void* const* d_in, const int* in_sizes, int n_in,
                              void* d_out, int out_size, void* d_ws,
                              size_t ws_size, hipStream_t stream) {
  const float* X    = (const float*)d_in[0];  // hidden_states [B,S,D]
  const float* mask = (const float*)d_in[1];  // attention_mask [B,1,1,S]
  const float* bias = (const float*)d_in[2];  // bias_matrix_chunk [S,S]
  const float* coef = (const float*)d_in[3];  // bias_coef [1]
  const float* Wq   = (const float*)d_in[4];
  const float* bq   = (const float*)d_in[5];
  const float* Wk   = (const float*)d_in[6];
  const float* bk   = (const float*)d_in[7];
  const float* Wv   = (const float*)d_in[8];
  const float* bv   = (const float*)d_in[9];
  float* out = (float*)d_out;

  // workspace partition (38 MB total)
  char* w = (char*)d_ws;
  _Float16* Xh  = (_Float16*)w; w += (size_t)NN * DD * sizeof(_Float16);
  _Float16* WTq = (_Float16*)w; w += (size_t)DD * DD * sizeof(_Float16);
  _Float16* WTk = (_Float16*)w; w += (size_t)DD * DD * sizeof(_Float16);
  _Float16* WTv = (_Float16*)w; w += (size_t)DD * DD * sizeof(_Float16);
  _Float16* Qh  = (_Float16*)w; w += (size_t)NN * DD * sizeof(_Float16);
  _Float16* Kh  = (_Float16*)w; w += (size_t)NN * DD * sizeof(_Float16);
  _Float16* Vt  = (_Float16*)w; w += (size_t)NN * DD * sizeof(_Float16);

  // Phase 0: precision conversion / weight transpose
  k_xcvt<<<dim3((NN * DD) / (256 * 4)), 256, 0, stream>>>(X, Xh);
  k_wtrans<<<dim3(DD / 32, DD / 32, 3), 256, 0, stream>>>(Wq, Wk, Wv, WTq, WTk,
                                                          WTv);
  // Phase 1: QKV projections (WMMA)
  k_qkv_gemm<<<dim3(NN / 32, HH, 3), 32, 0, stream>>>(Xh, WTq, WTk, WTv, bq, bk,
                                                      bv, Qh, Kh, Vt);
  // Phase 2: attention (double-buffered async staging + WMMA + online softmax)
  k_attn<<<dim3(SS / 64, HH, BB), 128, 0, stream>>>(Qh, Kh, Vt, bias, coef,
                                                    mask, out);
}
